// diffpool_layer_hetero_39410619908842
// MI455X (gfx1250) — compile-verified, matrix operand order checked by reading, not run on hardware
//
#include <hip/hip_runtime.h>
#include <hip/hip_bf16.h>

typedef __attribute__((ext_vector_type(16))) __bf16 v16bf;
typedef __attribute__((ext_vector_type(8)))  float  v8f;

#define ND   8192
#define NPR  8192
#define F    128      // F_IN == F_OUT == K_CL
#define BASES 30

// ---------------------------------------------------------------------------
// WMMA helpers (CDNA5 gfx1250: V_WMMA_F32_16X16X32_BF16, wave32)
// ---------------------------------------------------------------------------
__device__ __forceinline__ v8f wmma_bf16(v16bf a, v16bf b, v8f c) {
  return __builtin_amdgcn_wmma_f32_16x16x32_bf16(
      /*neg_a=*/false, a, /*neg_b=*/false, b,
      /*c_mod=*/(short)0, c, /*reuse_a=*/false, /*reuse_b=*/false);
}

// Load one 16x32 K-major operand tile (A or B) per ISA 7.12.2 layout:
// lane L: row = L&15, kbase = (L>>4)*8; elements j<8 -> k = kbase+j,
// j>=8 -> k = 16+kbase+(j-8). src points at (row0, k0); rs = row stride,
// ks = k stride (in elements). Converts to bf16 in-register.
template <typename T>
__device__ __forceinline__ v16bf load_kmajor_tile(const T* src, long rs, long ks) {
  int lane = threadIdx.x & 31;
  int ri   = lane & 15;
  int kb   = (lane >> 4) * 8;
  const T* p = src + (long)ri * rs;
  v16bf out;
#pragma unroll
  for (int j = 0; j < 8; ++j)
    out[j] = (__bf16)(float)p[(long)(kb + j) * ks];
#pragma unroll
  for (int j = 0; j < 8; ++j)
    out[8 + j] = (__bf16)(float)p[(long)(16 + kb + j) * ks];
  return out;
}

// ---------------------------------------------------------------------------
// Utility kernels
// ---------------------------------------------------------------------------
__global__ void k_zero(float* p, long n) {
  long i = (long)blockIdx.x * blockDim.x + threadIdx.x;
  long st = (long)gridDim.x * blockDim.x;
  for (; i < n; i += st) p[i] = 0.0f;
}

__global__ void k_cast_bf16(const float* __restrict__ x, __bf16* __restrict__ y, long n) {
  long i = (long)blockIdx.x * blockDim.x + threadIdx.x;
  long st = (long)gridDim.x * blockDim.x;
  for (; i < n; i += st) y[i] = (__bf16)x[i];
}

// W_r[i][o] = sum_b coeff[r,b] * w[b,i,o]; stored transposed+bf16 as
// WT[((r*2+head)*128 + o)*128 + i]
__global__ void k_relw(const float* __restrict__ w_e, const float* __restrict__ c_e,
                       const float* __restrict__ w_a, const float* __restrict__ c_a,
                       __bf16* __restrict__ WT) {
  int t = blockIdx.x * blockDim.x + threadIdx.x;
  if (t >= 5 * 2 * F * F) return;
  int i = t & 127, o = (t >> 7) & 127, head = (t >> 14) & 1, r = t >> 15;
  const float* w = head ? w_a : w_e;
  const float* c = head ? c_a : c_e;
  float acc = 0.f;
#pragma unroll 5
  for (int b = 0; b < BASES; ++b)
    acc += c[r * BASES + b] * w[((long)b * F + i) * F + o];
  WT[t] = (__bf16)acc;
}

// selfw [i][o] f32 -> ST[head][o][i] bf16
__global__ void k_selfwT(const float* __restrict__ se, const float* __restrict__ sa,
                         __bf16* __restrict__ ST) {
  int t = blockIdx.x * blockDim.x + threadIdx.x;
  if (t >= 2 * F * F) return;
  int i = t & 127, o = (t >> 7) & 127, head = t >> 14;
  const float* s = head ? sa : se;
  ST[t] = (__bf16)s[i * F + o];
}

// ---------------------------------------------------------------------------
// Edge scatter: one wave per edge, float4 per lane, L2 atomics
// ---------------------------------------------------------------------------
__global__ void k_scatter_mean(const float* __restrict__ x,
                               const int* __restrict__ src, const int* __restrict__ dst,
                               int E, float* __restrict__ Y, float* __restrict__ cnt) {
  int wid  = blockIdx.x * (blockDim.x >> 5) + (threadIdx.x >> 5);
  int lane = threadIdx.x & 31;
  if (wid >= E) return;
  int s = src[wid], d = dst[wid];
  float4 v = *(const float4*)(x + (long)s * F + lane * 4);
  float* yp = Y + (long)d * F + lane * 4;
  atomicAdd(yp + 0, v.x); atomicAdd(yp + 1, v.y);
  atomicAdd(yp + 2, v.z); atomicAdd(yp + 3, v.w);
  if (lane == 0) atomicAdd(cnt + d, 1.0f);
}

__global__ void k_scatter_sum(const float* __restrict__ x,
                              const int* __restrict__ src, const int* __restrict__ dst,
                              int E, float* __restrict__ P) {
  int wid  = blockIdx.x * (blockDim.x >> 5) + (threadIdx.x >> 5);
  int lane = threadIdx.x & 31;
  if (wid >= E) return;
  int s = src[wid], d = dst[wid];
  float4 v = *(const float4*)(x + (long)s * F + lane * 4);
  float* yp = P + (long)d * F + lane * 4;
  atomicAdd(yp + 0, v.x); atomicAdd(yp + 1, v.y);
  atomicAdd(yp + 2, v.z); atomicAdd(yp + 3, v.w);
}

__global__ void k_normcast(const float* __restrict__ Y, const float* __restrict__ cnt,
                           __bf16* __restrict__ Yb) {
  int t = blockIdx.x * blockDim.x + threadIdx.x;  // ND*F threads
  int n = t >> 7;
  float c = cnt[n];
  c = c < 1.0f ? 1.0f : c;
  Yb[t] = (__bf16)(Y[t] / c);
}

// ---------------------------------------------------------------------------
// GEMM 1: h_{emb,asg}[M x 128] += Ybf[M x 128] @ W_r (N=256 concat heads)
// one wave per 16x16 output tile; K=128 -> 4 WMMA
// ---------------------------------------------------------------------------
__global__ void k_gemm_h(const __bf16* __restrict__ A, const __bf16* __restrict__ BT,
                         float* __restrict__ h0, float* __restrict__ h1) {
  int wave = blockIdx.x * 4 + (threadIdx.x >> 5);
  int tn = wave & 15;       // 256/16 column tiles
  int tm = wave >> 4;       // 8192/16 row tiles
  int lane = threadIdx.x & 31;
  const __bf16* a0 = A  + (long)tm * 16 * F;
  const __bf16* b0 = BT + (long)tn * 16 * F;
  v8f acc = {};
#pragma unroll
  for (int kk = 0; kk < F; kk += 32) {
    v16bf a = load_kmajor_tile(a0 + kk, F, 1);
    v16bf b = load_kmajor_tile(b0 + kk, F, 1);
    acc = wmma_bf16(a, b, acc);
  }
  int col  = tn * 16 + (lane & 15);
  int rowb = tm * 16 + (lane >> 4) * 8;
  float* h = (col < F) ? h0 : h1;
  int c2 = col & 127;
#pragma unroll
  for (int j = 0; j < 8; ++j)
    h[(long)(rowb + j) * F + c2] += acc[j];
}

// ---------------------------------------------------------------------------
// GEMM 2: out_emb = relu(X @ selfw_e + h_emb + b_e); s = X @ selfw_a + h_asg + b_a
// ---------------------------------------------------------------------------
__global__ void k_gemm_self(const __bf16* __restrict__ A, const __bf16* __restrict__ ST,
                            const float* __restrict__ h_emb, const float* __restrict__ h_asg,
                            const float* __restrict__ bias_e, const float* __restrict__ bias_a,
                            float* __restrict__ out_emb, float* __restrict__ out_s) {
  int wave = blockIdx.x * 4 + (threadIdx.x >> 5);
  int tn = wave & 15;
  int tm = wave >> 4;
  int lane = threadIdx.x & 31;
  const __bf16* a0 = A  + (long)tm * 16 * F;
  const __bf16* b0 = ST + (long)tn * 16 * F;
  v8f acc = {};
#pragma unroll
  for (int kk = 0; kk < F; kk += 32) {
    v16bf a = load_kmajor_tile(a0 + kk, F, 1);
    v16bf b = load_kmajor_tile(b0 + kk, F, 1);
    acc = wmma_bf16(a, b, acc);
  }
  int col  = tn * 16 + (lane & 15);
  int rowb = tm * 16 + (lane >> 4) * 8;
  bool head1 = col >= F;
  int c2 = col & 127;
  const float* h  = head1 ? h_asg : h_emb;
  const float* bs = head1 ? bias_a : bias_e;
  float* o        = head1 ? out_s : out_emb;
  float bv = bs[c2];
#pragma unroll
  for (int j = 0; j < 8; ++j) {
    long idx = (long)(rowb + j) * F + c2;
    float v = acc[j] + h[idx] + bv;
    if (!head1) v = fmaxf(v, 0.0f);
    o[idx] = v;
  }
}

// ---------------------------------------------------------------------------
// GEMM 3: out[128x128] += S^T @ P  (K = 8192, split-K, atomic epilogue)
// A(m,k) = S[k*128+m], B(n,k) = P[k*128+n]; 8x8 tiles x 16 K-chunks of 512
// ---------------------------------------------------------------------------
__global__ void k_gemm_tn(const float* __restrict__ S, const float* __restrict__ P,
                          float* __restrict__ out) {
  int wave = blockIdx.x * 4 + (threadIdx.x >> 5);
  int tn = wave & 7;
  int tm = (wave >> 3) & 7;
  int ch = wave >> 6;            // 0..15
  long k0 = (long)ch * 512;
  v8f acc = {};
#pragma unroll 4
  for (int kk = 0; kk < 512; kk += 32) {
    v16bf a = load_kmajor_tile(S + (k0 + kk) * F + tm * 16, 1, F);
    v16bf b = load_kmajor_tile(P + (k0 + kk) * F + tn * 16, 1, F);
    acc = wmma_bf16(a, b, acc);
  }
  int lane = threadIdx.x & 31;
  int col  = tn * 16 + (lane & 15);
  int rowb = tm * 16 + (lane >> 4) * 8;
#pragma unroll
  for (int j = 0; j < 8; ++j)
    atomicAdd(out + (long)(rowb + j) * F + col, acc[j]);
}

// ---------------------------------------------------------------------------
extern "C" void kernel_launch(void* const* d_in, const int* in_sizes, int n_in,
                              void* d_out, int out_size, void* d_ws, size_t ws_size,
                              hipStream_t stream) {
  const float* x_d = (const float*)d_in[0];
  const float* x_p = (const float*)d_in[1];
  const int* e_src[5] = {(const int*)d_in[2], (const int*)d_in[4], (const int*)d_in[6],
                         (const int*)d_in[8], (const int*)d_in[10]};
  const int* e_dst[5] = {(const int*)d_in[3], (const int*)d_in[5], (const int*)d_in[7],
                         (const int*)d_in[9], (const int*)d_in[11]};
  const int  e_cnt[5] = {in_sizes[2], in_sizes[4], in_sizes[6], in_sizes[8], in_sizes[10]};
  const float* w_e  = (const float*)d_in[12];
  const float* c_e  = (const float*)d_in[13];
  const float* sw_e = (const float*)d_in[14];
  const float* b_e  = (const float*)d_in[15];
  const float* w_a  = (const float*)d_in[16];
  const float* c_a  = (const float*)d_in[17];
  const float* sw_a = (const float*)d_in[18];
  const float* b_a  = (const float*)d_in[19];

  // workspace carve (256B aligned slices)
  char* ws = (char*)d_ws;
  auto carve = [&](size_t bytes) -> void* {
    void* p = (void*)ws;
    ws += (bytes + 255) & ~(size_t)255;
    return p;
  };
  __bf16* WT  = (__bf16*)carve(5 * 2 * F * F * sizeof(__bf16));
  __bf16* ST  = (__bf16*)carve(2 * F * F * sizeof(__bf16));
  __bf16* xdb = (__bf16*)carve((size_t)ND  * F * sizeof(__bf16));
  __bf16* xpb = (__bf16*)carve((size_t)NPR * F * sizeof(__bf16));
  float*  Y   = (float*)carve((size_t)ND * F * sizeof(float));    // also reused as P
  float*  cnt = (float*)carve(ND * sizeof(float));                 // directly after Y
  __bf16* Yb  = (__bf16*)carve((size_t)ND * F * sizeof(__bf16));
  float*  hbase = (float*)carve((size_t)4 * ND * F * sizeof(float));
  float*  h_emb_d = hbase + 0L * ND * F;
  float*  h_asg_d = hbase + 1L * ND * F;
  float*  h_emb_p = hbase + 2L * ND * F;
  float*  h_asg_p = hbase + 3L * ND * F;
  float*  emb_d = (float*)carve((size_t)ND  * F * sizeof(float));
  float*  emb_p = (float*)carve((size_t)NPR * F * sizeof(float));

  float* adj2 = (float*)d_out;                 // [128 x 128]
  float* emb2 = adj2 + F * F;                  // [128 x 128]
  float* s_d  = adj2 + 2 * F * F;              // [8192 x 128]
  float* s_p  = s_d + (long)ND * F;            // [8192 x 128]

  const long NF = (long)ND * F;

  // 0) zero accumulators
  k_zero<<<1024, 256, 0, stream>>>(hbase, 4 * NF);
  k_zero<<<64,   256, 0, stream>>>(adj2, 2L * F * F);

  // 1) per-relation weights, self-weight transpose, bf16 inputs
  k_relw<<<(5 * 2 * F * F + 255) / 256, 256, 0, stream>>>(w_e, c_e, w_a, c_a, WT);
  k_selfwT<<<(2 * F * F + 255) / 256, 256, 0, stream>>>(sw_e, sw_a, ST);
  k_cast_bf16<<<1024, 256, 0, stream>>>(x_d, xdb, NF);
  k_cast_bf16<<<1024, 256, 0, stream>>>(x_p, xpb, NF);

  // 2) RGCN message passing: etype r -> (src feats, dst h buffers)
  const float* r_x[5]  = {x_d, x_d, x_d, x_p, x_p};
  float* r_he[5] = {h_emb_d, h_emb_d, h_emb_p, h_emb_d, h_emb_p};
  float* r_ha[5] = {h_asg_d, h_asg_d, h_asg_p, h_asg_d, h_asg_p};
  for (int r = 0; r < 5; ++r) {
    k_zero<<<1024, 256, 0, stream>>>(Y, NF + ND);  // Y and cnt (contiguous)
    int E = e_cnt[r];
    k_scatter_mean<<<(E + 7) / 8, 256, 0, stream>>>(r_x[r], e_src[r], e_dst[r], E, Y, cnt);
    k_normcast<<<(int)(NF / 256), 256, 0, stream>>>(Y, cnt, Yb);
    k_gemm_h<<<(ND / 16) * 16 / 4, 128, 0, stream>>>(Yb, WT + (long)r * 2 * F * F,
                                                     r_he[r], r_ha[r]);
  }

  // 3) self-loop + bias (+relu for embed head); s_d/s_p straight into d_out
  k_gemm_self<<<(ND / 16) * 16 / 4, 128, 0, stream>>>(xdb, ST, h_emb_d, h_asg_d,
                                                      b_e, b_a, emb_d, s_d);
  k_gemm_self<<<(ND / 16) * 16 / 4, 128, 0, stream>>>(xpb, ST, h_emb_p, h_asg_p,
                                                      b_e, b_a, emb_p, s_p);

  // 4) pooling: adj2 += s_dst^T @ (A @ s_src) per adjacency, via scatter + split-K GEMM
  float* P = Y;  // reuse
  // ddi (two etypes accumulate into one adjacency)
  k_zero<<<1024, 256, 0, stream>>>(P, NF);
  k_scatter_sum<<<(e_cnt[0] + 7) / 8, 256, 0, stream>>>(s_d, e_src[0], e_dst[0], e_cnt[0], P);
  k_scatter_sum<<<(e_cnt[1] + 7) / 8, 256, 0, stream>>>(s_d, e_src[1], e_dst[1], e_cnt[1], P);
  k_gemm_tn<<<256, 128, 0, stream>>>(s_d, P, adj2);
  // ppi
  k_zero<<<1024, 256, 0, stream>>>(P, NF);
  k_scatter_sum<<<(e_cnt[4] + 7) / 8, 256, 0, stream>>>(s_p, e_src[4], e_dst[4], e_cnt[4], P);
  k_gemm_tn<<<256, 128, 0, stream>>>(s_p, P, adj2);
  // pdi: src = protein, dst = drug
  k_zero<<<1024, 256, 0, stream>>>(P, NF);
  k_scatter_sum<<<(e_cnt[3] + 7) / 8, 256, 0, stream>>>(s_p, e_src[3], e_dst[3], e_cnt[3], P);
  k_gemm_tn<<<256, 128, 0, stream>>>(s_d, P, adj2);

  // 5) embedding2 = s_d^T @ emb_d + s_p^T @ emb_p
  k_gemm_tn<<<256, 128, 0, stream>>>(s_d, emb_d, emb2);
  k_gemm_tn<<<256, 128, 0, stream>>>(s_p, emb_p, emb2);
}
